// NoisyMixtureOfExperts1_71536975282233
// MI455X (gfx1250) — compile-verified, matrix operand order checked by reading, not run on hardware
//
#include <hip/hip_runtime.h>
#include <hip/hip_bf16.h>

// ---------------------------------------------------------------------------
// NoisyMixtureOfExperts on gfx1250 (MI455X)
//   out[n,f] = sum_e coef[n,e] * (x[n,:] @ W_e[:,f] + b_e[f])
//   coef[n,e] = softmax(x@gw.T + gb + 0.1*noise)[n,e] if e in top-2 else 0
// Dense formulation -> bf16 WMMA GEMM with per-row/per-expert A scaling.
// B tiles staged via the Tensor Data Mover (tensor_load_to_lds, 6-arg
// clang-23 form); sync-staging fallback if the builtin is absent.
// ---------------------------------------------------------------------------

typedef __attribute__((ext_vector_type(16))) __bf16 v16bf;
typedef __attribute__((ext_vector_type(8)))  float  v8f;
typedef __attribute__((ext_vector_type(4)))  unsigned int u32x4;
typedef __attribute__((ext_vector_type(8)))  int          i32x8;
typedef __attribute__((ext_vector_type(4)))  int          i32x4;

#define NTOK 16384
#define DDIM 768
#define NEXP 8
#define FDIM 768
#define BM 128
#define BN 128
#define BK 32
#define KB_PER_E (DDIM / BK)       // 24
#define TOTAL_KB (NEXP * KB_PER_E) // 192
#define LDS_STRIDE 40              // bf16/row: 80 B pitch, 16B-aligned, conflict-free

#if __has_builtin(__builtin_amdgcn_tensor_load_to_lds)
#define USE_TDM 1
#else
#define USE_TDM 0
#endif

// float -> bf16 round-to-nearest-even
__device__ __forceinline__ unsigned short f2bf(float f) {
    unsigned int u = __float_as_uint(f);
    unsigned int r = u + 0x7fffu + ((u >> 16) & 1u);
    return (unsigned short)(r >> 16);
}

// packed bf16 multiply via CDNA5 VOP3P
__device__ __forceinline__ unsigned int pk_mul_bf16(unsigned int a, unsigned int b) {
    unsigned int d;
    asm("v_pk_mul_bf16 %0, %1, %2" : "=v"(d) : "v"(a), "v"(b));
    return d;
}

// generic LDS pointer -> LDS byte offset (ptrtoint of addrspace(3))
__device__ __forceinline__ unsigned int lds_offset_of(const void* p) {
    return (unsigned int)(unsigned long long)
        (__attribute__((address_space(3))) const void*)p;
}

// -------------------------- 1) gating ---------------------------------------
__global__ void gate_kernel(const float* __restrict__ x,
                            const float* __restrict__ noise,
                            const float* __restrict__ gw,
                            const float* __restrict__ gb,
                            float* __restrict__ coef) {
    int t = threadIdx.x;
    int lane = t & 31, wid = t >> 5;
    int e = lane & 7;
    int token = blockIdx.x * 32 + (wid << 2) + (lane >> 3);

    const float4* xr = (const float4*)(x + (size_t)token * DDIM);
    const float4* wr = (const float4*)(gw + (size_t)e * DDIM);
    float acc = 0.f;
#pragma unroll 4
    for (int i = 0; i < DDIM / 4; ++i) {
        float4 a = xr[i], b = wr[i];
        acc += a.x * b.x + a.y * b.y + a.z * b.z + a.w * b.w;
    }
    float score = acc + gb[e] + noise[(size_t)token * NEXP + e] * 0.1f;

    float m = score;
    for (int o = 4; o; o >>= 1) m = fmaxf(m, __shfl_xor(m, o, 8));
    float ex = expf(score - m);
    float s = ex;
    for (int o = 4; o; o >>= 1) s += __shfl_xor(s, o, 8);
    float w = ex / s;

    float v1 = w; int i1 = e;
    for (int o = 4; o; o >>= 1) {
        float ov = __shfl_xor(v1, o, 8);
        int   oi = __shfl_xor(i1, o, 8);
        if (ov > v1 || (ov == v1 && oi < i1)) { v1 = ov; i1 = oi; }
    }
    float v2 = (e == i1) ? -1.0f : w; int i2 = e;
    for (int o = 4; o; o >>= 1) {
        float ov = __shfl_xor(v2, o, 8);
        int   oi = __shfl_xor(i2, o, 8);
        if (ov > v2 || (ov == v2 && oi < i2)) { v2 = ov; i2 = oi; }
    }
    coef[(size_t)token * NEXP + e] = (e == i1 || e == i2) ? w : 0.0f;
}

// -------------------------- 2) x -> bf16 ------------------------------------
__global__ void cast_x_kernel(const float* __restrict__ x,
                              unsigned short* __restrict__ xb, int n4) {
    int i = blockIdx.x * blockDim.x + threadIdx.x;
    if (i >= n4) return;
    float4 v = ((const float4*)x)[i];
    ushort4 o;
    o.x = f2bf(v.x); o.y = f2bf(v.y); o.z = f2bf(v.z); o.w = f2bf(v.w);
    ((ushort4*)xb)[i] = o;
}

// -------------------------- 3) W[e,k,f] -> bf16 Wt[e,f,k] -------------------
__global__ void transpose_w_kernel(const float* __restrict__ w,
                                   unsigned short* __restrict__ wt) {
    __shared__ unsigned short tile[32][33];
    int e = blockIdx.z;
    int k0 = blockIdx.y * 32, f0 = blockIdx.x * 32;
    int t = threadIdx.x;
    int fl = t & 31, kq = t >> 5;
#pragma unroll
    for (int j = 0; j < 4; ++j) {
        int k = kq * 4 + j;
        tile[k][fl] = f2bf(w[((size_t)e * DDIM + k0 + k) * FDIM + f0 + fl]);
    }
    __syncthreads();
    int kl = t & 31, fq = t >> 5;
#pragma unroll
    for (int j = 0; j < 4; ++j) {
        int f = fq * 4 + j;
        wt[((size_t)e * FDIM + f0 + f) * DDIM + k0 + kl] = tile[kl][f];
    }
}

// -------------------------- 4) fused MoE GEMM -------------------------------
__device__ __forceinline__ v16bf load_frag(const unsigned short* base, int lane) {
    int row = lane & 15;
    int ko = (lane >> 4) << 3;
    const unsigned short* p = base + row * LDS_STRIDE + ko;
    union { uint4 u[2]; v16bf v; } u;
    u.u[0] = *(const uint4*)(p);
    u.u[1] = *(const uint4*)(p + 16);
    return u.v;
}

#if USE_TDM
// One TDM descriptor moves the whole 32(k) x 128(f) bf16 B tile into a padded
// LDS layout: 64B of data per line + 16B pad  ==  LDS_STRIDE(40) * 2B pitch.
__device__ __forceinline__ void tdm_load_b_tile(const unsigned short* gsrc,
                                                unsigned short* lds_dst) {
    unsigned long long ga = (unsigned long long)gsrc;
    unsigned int lofs = lds_offset_of(lds_dst);

    u32x4 g0;
    g0.x = 1u;                                  // count=1, user descriptor
    g0.y = lofs;                                // lds_addr (bytes)
    g0.z = (unsigned int)(ga & 0xffffffffu);    // global_addr[31:0]
    g0.w = (unsigned int)((ga >> 32) & 0x01ffffffu) | (2u << 30); // [56:32], type=2

    i32x8 g1;
    g1[0] = (1 << 16)      // data_size = 2 bytes
          | (1 << 20)      // pad_enable
          | (3 << 22)      // pad_interval: 16 DWORDs (= 64 B) of data per line
          | (3 << 25);     // pad_amount:   4 DWORDs (= 16 B) padding
    g1[1] = (int)(((unsigned)BK & 0xffffu) << 16);          // tensor_dim0 lo16
    g1[2] = (int)(((unsigned)BN & 0xffffu) << 16);          // dim0 hi=0 | tensor_dim1 lo16
    g1[3] = (int)(((unsigned)BK & 0xffffu) << 16);          // dim1 hi=0 | tile_dim0=32
    g1[4] = (int)((unsigned)BN & 0xffffu);                  // tile_dim1=128, tile_dim2=0
    g1[5] = (int)DDIM;                                      // tensor_dim0_stride lo32 = 768
    g1[6] = 0;                                              // stride hi | dim1_stride lo
    g1[7] = 0;

    i32x4 z4 = {0, 0, 0, 0};
    i32x8 z8 = {0, 0, 0, 0, 0, 0, 0, 0};
    // clang-23 / therock-10.0 form: (g0, g1, g2, g3, extra, cpol)
    __builtin_amdgcn_tensor_load_to_lds(g0, g1, z4, z4, z8, 0);
}

__device__ __forceinline__ void tdm_wait0() {
#if __has_builtin(__builtin_amdgcn_s_wait_tensorcnt)
    __builtin_amdgcn_s_wait_tensorcnt(0);
#else
    asm volatile("s_wait_tensorcnt 0x0" ::: "memory");
#endif
}
#endif

__global__ __launch_bounds__(256) void moe_gemm_kernel(
    const unsigned short* __restrict__ xb,   // [N, D] bf16
    const unsigned short* __restrict__ wt,   // [E, F, D] bf16 (pre-transposed)
    const float* __restrict__ coef,          // [N, E]
    const float* __restrict__ eb,            // [E, F]
    float* __restrict__ out) {               // [N, F]
    __shared__ float coef_s[BM * NEXP];
    __shared__ __align__(16) unsigned short As[2][BM * LDS_STRIDE];
    __shared__ __align__(16) unsigned short Bs[2][BN * LDS_STRIDE];

    const int t = threadIdx.x;
    const int lane = t & 31, wid = t >> 5;
    const int wm = wid >> 1, wn = wid & 1;   // 4x2 wave grid: 32x64 per wave
    const int m_base = blockIdx.x * BM;
    const int f_base = blockIdx.y * BN;

    for (int i = t; i < BM * NEXP; i += 256)
        coef_s[i] = coef[(size_t)m_base * NEXP + i];
    __syncthreads();

    const int r = t >> 1;
    const int half = t & 1;

    auto stage_a = [&](int idx, int buf) {
        int e = idx / KB_PER_E, kb = idx % KB_PER_E;
        const unsigned short* src =
            xb + (size_t)(m_base + r) * DDIM + kb * BK + half * 16;
        uint4 p0 = *(const uint4*)(src);
        uint4 p1 = *(const uint4*)(src + 8);
        unsigned short cb = f2bf(coef_s[r * NEXP + e]);
        unsigned int cpk = (unsigned int)cb | ((unsigned int)cb << 16);
        p0.x = pk_mul_bf16(p0.x, cpk); p0.y = pk_mul_bf16(p0.y, cpk);
        p0.z = pk_mul_bf16(p0.z, cpk); p0.w = pk_mul_bf16(p0.w, cpk);
        p1.x = pk_mul_bf16(p1.x, cpk); p1.y = pk_mul_bf16(p1.y, cpk);
        p1.z = pk_mul_bf16(p1.z, cpk); p1.w = pk_mul_bf16(p1.w, cpk);
        unsigned short* dst = &As[buf][r * LDS_STRIDE + half * 16];
        *(uint4*)(dst) = p0;
        *(uint4*)(dst + 8) = p1;
    };

    auto stage_b = [&](int idx, int buf) {
        int e = idx / KB_PER_E, kb = idx % KB_PER_E;
        const unsigned short* src =
            wt + ((size_t)e * FDIM + f_base) * DDIM + kb * BK;
#if USE_TDM
        if (wid == 0) tdm_load_b_tile(src, &Bs[buf][0]);
#else
        const unsigned short* s2 = src + (size_t)r * DDIM + half * 16;
        uint4 q0 = *(const uint4*)(s2);
        uint4 q1 = *(const uint4*)(s2 + 8);
        unsigned short* dst = &Bs[buf][r * LDS_STRIDE + half * 16];
        *(uint4*)(dst) = q0;
        *(uint4*)(dst + 8) = q1;
#endif
    };

    v8f c[2][4] = {};
    stage_a(0, 0);
    stage_b(0, 0);

    for (int idx = 0; idx < TOTAL_KB; ++idx) {
#if USE_TDM
        if (wid == 0) tdm_wait0();   // TDM write of buffer idx&1 complete
#endif
        __syncthreads();
        if (idx + 1 < TOTAL_KB) {
            stage_a(idx + 1, (idx + 1) & 1);
            stage_b(idx + 1, (idx + 1) & 1);
        }
        const int buf = idx & 1;

        v16bf a0 = load_frag(&As[buf][(wm * 32 + 0)  * LDS_STRIDE], lane);
        v16bf a1 = load_frag(&As[buf][(wm * 32 + 16) * LDS_STRIDE], lane);
        v16bf b0 = load_frag(&Bs[buf][(wn * 64 + 0)  * LDS_STRIDE], lane);
        v16bf b1 = load_frag(&Bs[buf][(wn * 64 + 16) * LDS_STRIDE], lane);
        v16bf b2 = load_frag(&Bs[buf][(wn * 64 + 32) * LDS_STRIDE], lane);
        v16bf b3 = load_frag(&Bs[buf][(wn * 64 + 48) * LDS_STRIDE], lane);

        c[0][0] = __builtin_amdgcn_wmma_f32_16x16x32_bf16(false, a0, false, b0, (short)0, c[0][0], false, false);
        c[0][1] = __builtin_amdgcn_wmma_f32_16x16x32_bf16(false, a0, false, b1, (short)0, c[0][1], false, false);
        c[0][2] = __builtin_amdgcn_wmma_f32_16x16x32_bf16(false, a0, false, b2, (short)0, c[0][2], false, false);
        c[0][3] = __builtin_amdgcn_wmma_f32_16x16x32_bf16(false, a0, false, b3, (short)0, c[0][3], false, false);
        c[1][0] = __builtin_amdgcn_wmma_f32_16x16x32_bf16(false, a1, false, b0, (short)0, c[1][0], false, false);
        c[1][1] = __builtin_amdgcn_wmma_f32_16x16x32_bf16(false, a1, false, b1, (short)0, c[1][1], false, false);
        c[1][2] = __builtin_amdgcn_wmma_f32_16x16x32_bf16(false, a1, false, b2, (short)0, c[1][2], false, false);
        c[1][3] = __builtin_amdgcn_wmma_f32_16x16x32_bf16(false, a1, false, b3, (short)0, c[1][3], false, false);
    }
    __syncthreads();

    // epilogue: + sum_e coef[row,e] * b_e[f], store fp32
#pragma unroll
    for (int mt = 0; mt < 2; ++mt) {
        int rbase = wm * 32 + mt * 16 + ((lane >> 4) << 3);
#pragma unroll
        for (int nt = 0; nt < 4; ++nt) {
            int f = f_base + wn * 64 + nt * 16 + (lane & 15);
            float bv[NEXP];
#pragma unroll
            for (int e = 0; e < NEXP; ++e) bv[e] = eb[e * FDIM + f];
#pragma unroll
            for (int i = 0; i < 8; ++i) {
                int rl = rbase + i;
                const float* cr = &coef_s[rl * NEXP];
                float bias = cr[0] * bv[0] + cr[1] * bv[1] + cr[2] * bv[2] + cr[3] * bv[3]
                           + cr[4] * bv[4] + cr[5] * bv[5] + cr[6] * bv[6] + cr[7] * bv[7];
                out[(size_t)(m_base + rl) * FDIM + f] = c[mt][nt][i] + bias;
            }
        }
    }
}

// ---------------------------------------------------------------------------
extern "C" void kernel_launch(void* const* d_in, const int* in_sizes, int n_in,
                              void* d_out, int out_size, void* d_ws, size_t ws_size,
                              hipStream_t stream) {
    const float* x        = (const float*)d_in[0];
    const float* noise    = (const float*)d_in[1];
    const float* gate_w   = (const float*)d_in[2];
    const float* gate_b   = (const float*)d_in[3];
    const float* expert_w = (const float*)d_in[4];
    const float* expert_b = (const float*)d_in[5];
    float* out = (float*)d_out;

    // workspace: coef[N,8] f32 | xb[N,768] bf16 | wt[8,768,768] bf16  (~35 MB)
    float* coef = (float*)d_ws;
    unsigned short* xb = (unsigned short*)(coef + (size_t)NTOK * NEXP);
    unsigned short* wt = xb + (size_t)NTOK * DDIM;

    gate_kernel<<<NTOK / 32, 256, 0, stream>>>(x, noise, gate_w, gate_b, coef);
    cast_x_kernel<<<(NTOK * DDIM / 4 + 255) / 256, 256, 0, stream>>>(x, xb, NTOK * DDIM / 4);
    transpose_w_kernel<<<dim3(FDIM / 32, DDIM / 32, NEXP), 256, 0, stream>>>(expert_w, wt);
    moe_gemm_kernel<<<dim3(NTOK / BM, FDIM / BN), 256, 0, stream>>>(xb, wt, coef, expert_b, out);
}
